// Model_13271448945168
// MI455X (gfx1250) — compile-verified, matrix operand order checked by reading, not run on hardware
//
#include <hip/hip_runtime.h>
#include <cstdint>

#define IN_DIM 128
#define HID    64
#define W2     128
#define ROWS   32     // rows of the node matrix processed per block in GEMM kernels
#define LDK128 136    // padded LDS stride (halves) for K=128 tiles (+8 avoids bank conflicts)
#define LDK64  72     // padded LDS stride (halves) for K=64 tiles

typedef __attribute__((ext_vector_type(16))) _Float16 v16h;
typedef __attribute__((ext_vector_type(8)))  _Float16 v8h;
typedef __attribute__((ext_vector_type(8)))  float    v8f;

// ---------------------------------------------------------------------------
// WMMA fragment helper (16x16x32 f16). Per CDNA5 ISA the 16 halves
// a lane holds are two contiguous K-runs of 8:
//   lanes 0-15 : K = k0+[0..7] and k0+[16..23]   (row = r0 + lane)
//   lanes 16-31: K = k0+[8..15] and k0+[24..31]  (row = r0 + lane-16)
// With tiles stored K-contiguous (A row-major, B transposed/N-major) each
// fragment is exactly two ds_load_b128.
// ---------------------------------------------------------------------------
__device__ __forceinline__ v16h frag_row(const _Float16* __restrict__ base, int ld,
                                         int r0, int k0, int lane) {
  int r  = r0 + (lane & 15);
  int kb = k0 + ((lane >> 4) << 3);
  const _Float16* p = base + r * ld + kb;
  v8h lo = *(const v8h*)p;          // K = kb .. kb+7
  v8h hi = *(const v8h*)(p + 16);   // K = kb+16 .. kb+23
  return __builtin_shufflevector(lo, hi, 0, 1, 2, 3, 4, 5, 6, 7,
                                         8, 9, 10, 11, 12, 13, 14, 15);
}

__device__ __forceinline__ v8f wmma_f16(v16h a, v16h b, v8f c) {
  // 8 args: (neg_a, A, neg_b, B, c_mod, C, reuse_a, reuse_b)
  return __builtin_amdgcn_wmma_f32_16x16x32_f16(false, a, false, b, (short)0, c, false, false);
}

// ---------------------------------------------------------------------------
// Generic fill
// ---------------------------------------------------------------------------
__global__ void fill_kernel(float* __restrict__ p, size_t n, float v) {
  size_t i = (size_t)blockIdx.x * blockDim.x + threadIdx.x;
  if (i < n) p[i] = v;
}

// ---------------------------------------------------------------------------
// Fused input MLP:  h3 = leaky(h@Win+bin @Wt1+bt1)@Wt2+bt2, plus gate logits.
// 32 rows/block, 8 waves -> 2x4 grid of 16x16 tiles, WMMA chained in LDS.
// Weights staged transposed (N-major, K-contiguous, padded stride).
// ---------------------------------------------------------------------------
__global__ void __launch_bounds__(256) fused_mlp_kernel(
    const float* __restrict__ h,   const float* __restrict__ Win, const float* __restrict__ bin,
    const float* __restrict__ Wt1, const float* __restrict__ bt1,
    const float* __restrict__ Wt2, const float* __restrict__ bt2,
    const float* __restrict__ Wgate, const float* __restrict__ bgate,
    float* __restrict__ h3, float* __restrict__ logits, int N)
{
  __shared__ __align__(16) _Float16 sWinT[HID * LDK128];  // [n][k] k<128
  __shared__ __align__(16) _Float16 sWt1T[HID * LDK64];   // [n][k] k<64
  __shared__ __align__(16) _Float16 sWt2T[HID * LDK64];
  __shared__ __align__(16) _Float16 sA  [ROWS * LDK128];  // [m][k]
  __shared__ __align__(16) _Float16 sTa [ROWS * LDK64];
  __shared__ __align__(16) _Float16 sTb [ROWS * LDK64];
  __shared__ float sH3[ROWS * HID];

  const int tid  = threadIdx.x;
  const int lane = tid & 31;
  const int w    = tid >> 5;
  const int row0 = blockIdx.x * ROWS;

  for (int i = tid; i < IN_DIM * HID; i += 256) {          // Win[k][c] -> sWinT[c][k]
    int k = i >> 6, c = i & 63;
    sWinT[c * LDK128 + k] = (_Float16)Win[i];
  }
  for (int i = tid; i < HID * HID; i += 256) {             // Wt*[k][c] -> sW*T[c][k]
    int k = i >> 6, c = i & 63;
    sWt1T[c * LDK64 + k] = (_Float16)Wt1[i];
    sWt2T[c * LDK64 + k] = (_Float16)Wt2[i];
  }
  for (int i = tid; i < ROWS * IN_DIM; i += 256) {
    int r = i >> 7, c = i & 127;
    int gr = row0 + r;
    sA[r * LDK128 + c] = (gr < N) ? (_Float16)h[(size_t)gr * IN_DIM + c] : (_Float16)0.0f;
  }
  __syncthreads();

  const int tm = w >> 2, tn = w & 3;
  const int m0 = tm * 16, n0 = tn * 16;
  const int n  = n0 + (lane & 15);
  const int mh = m0 + ((lane >> 4) << 3);

  { // stage 1: h @ Win + bin
    v8f c = {};
#pragma unroll
    for (int k0 = 0; k0 < IN_DIM; k0 += 32)
      c = wmma_f16(frag_row(sA, LDK128, m0, k0, lane),
                   frag_row(sWinT, LDK128, n0, k0, lane), c);
    float bias = bin[n];
#pragma unroll
    for (int r = 0; r < 8; ++r) sTa[(mh + r) * LDK64 + n] = (_Float16)(c[r] + bias);
  }
  __syncthreads();

  { // stage 2: @ Wt1 + bt1, LeakyReLU(0.05)
    v8f c = {};
#pragma unroll
    for (int k0 = 0; k0 < HID; k0 += 32)
      c = wmma_f16(frag_row(sTa, LDK64, m0, k0, lane),
                   frag_row(sWt1T, LDK64, n0, k0, lane), c);
    float bias = bt1[n];
#pragma unroll
    for (int r = 0; r < 8; ++r) {
      float v = c[r] + bias;
      v = (v >= 0.0f) ? v : 0.05f * v;
      sTb[(mh + r) * LDK64 + n] = (_Float16)v;
    }
  }
  __syncthreads();

  { // stage 3: @ Wt2 + bt2 -> h3
    v8f c = {};
#pragma unroll
    for (int k0 = 0; k0 < HID; k0 += 32)
      c = wmma_f16(frag_row(sTb, LDK64, m0, k0, lane),
                   frag_row(sWt2T, LDK64, n0, k0, lane), c);
    float bias = bt2[n];
#pragma unroll
    for (int r = 0; r < 8; ++r) {
      float v = c[r] + bias;
      sH3[(mh + r) * HID + n] = v;
      int gr = row0 + mh + r;
      if (gr < N) h3[(size_t)gr * HID + n] = v;
    }
  }
  __syncthreads();

  // attention-gate logits: one thread per row
  if (tid < ROWS) {
    int gr = row0 + tid;
    if (gr < N) {
      float acc = bgate[0];
      for (int c = 0; c < HID; ++c) acc += sH3[tid * HID + c] * Wgate[c];
      logits[gr] = acc;
    }
  }
}

// ---------------------------------------------------------------------------
// Degrees: one thread per edge
// ---------------------------------------------------------------------------
__global__ void degree_kernel(const int* __restrict__ src, const int* __restrict__ dst,
                              float* __restrict__ outdeg, float* __restrict__ indeg, int E) {
  int e = blockIdx.x * blockDim.x + threadIdx.x;
  if (e < E) {
    atomicAdd(&outdeg[src[e]], 1.0f);
    atomicAdd(&indeg [dst[e]], 1.0f);
  }
}

// ---------------------------------------------------------------------------
// agg[dst] += h3[src] * rsqrt(outdeg[src]); 16 lanes / edge, float4 per lane
// ---------------------------------------------------------------------------
__global__ void gcn_scatter_kernel(const int* __restrict__ src, const int* __restrict__ dst,
                                   const float* __restrict__ h3, const float* __restrict__ outdeg,
                                   float* __restrict__ agg, int E) {
  int t = blockIdx.x * 256 + threadIdx.x;
  int e = t >> 4;
  int c = (t & 15) * 4;
  if (e >= E) return;
  int s = src[e], d = dst[e];
  float scale = rsqrtf(fmaxf(outdeg[s], 1.0f));
  const float4 v = *(const float4*)&h3[(size_t)s * HID + c];
  float* a = &agg[(size_t)d * HID + c];
  atomicAdd(a + 0, v.x * scale);
  atomicAdd(a + 1, v.y * scale);
  atomicAdd(a + 2, v.z * scale);
  atomicAdd(a + 3, v.w * scale);
}

// ---------------------------------------------------------------------------
// Softmax pooling reductions
// ---------------------------------------------------------------------------
__global__ void logit_max_kernel(const float* __restrict__ logits, float* __restrict__ maxbuf, int N) {
  __shared__ float s[256];
  float m = -3.4e38f;
  for (int i = threadIdx.x; i < N; i += 256) m = fmaxf(m, logits[i]);
  s[threadIdx.x] = m;
  __syncthreads();
  for (int st = 128; st > 0; st >>= 1) {
    if (threadIdx.x < st) s[threadIdx.x] = fmaxf(s[threadIdx.x], s[threadIdx.x + st]);
    __syncthreads();
  }
  if (threadIdx.x == 0) maxbuf[0] = s[0];
}

__global__ void pool_partial_kernel(const float* __restrict__ logits, const float* __restrict__ h3,
                                    const float* __restrict__ maxbuf,
                                    float* __restrict__ poolP, float* __restrict__ poolS, int N) {
  const int tid = threadIdx.x;
  const int c   = tid & 63;
  const int rs  = tid >> 6;          // 4 row-groups of 64 channel lanes
  const float mx = maxbuf[0];
  float accP = 0.f, accS = 0.f;
  for (int row = blockIdx.x * 4 + rs; row < N; row += gridDim.x * 4) {
    float e = __expf(logits[row] - mx);
    accP += e * h3[(size_t)row * HID + c];
    if (c == 0) accS += e;
  }
  __shared__ float sp[256];
  sp[tid] = accP;
  __shared__ float ss[4];
  if (c == 0) ss[rs] = accS;
  __syncthreads();
  if (tid < 64) atomicAdd(&poolP[tid], sp[tid] + sp[tid + 64] + sp[tid + 128] + sp[tid + 192]);
  if (tid == 0) atomicAdd(poolS, ss[0] + ss[1] + ss[2] + ss[3]);
}

// ---------------------------------------------------------------------------
// x = concat[ LN(agg*rsqrt(indeg) @ Wgcn + bgcn) - h3 ,  pool - h3 ]
// ---------------------------------------------------------------------------
__global__ void __launch_bounds__(256) build_x_kernel(
    const float* __restrict__ agg, const float* __restrict__ indeg,
    const float* __restrict__ Wgcn, const float* __restrict__ bgcn,
    const float* __restrict__ ggcn, const float* __restrict__ begcn,
    const float* __restrict__ h3,
    const float* __restrict__ poolP, const float* __restrict__ poolS,
    float* __restrict__ x, int N)
{
  __shared__ __align__(16) _Float16 sWT[HID * LDK64];   // [n][k]
  __shared__ __align__(16) _Float16 sA [ROWS * LDK64];  // [m][k]
  __shared__ float sY[ROWS * HID];
  __shared__ float spool[HID];

  const int tid = threadIdx.x, lane = tid & 31, w = tid >> 5;
  const int row0 = blockIdx.x * ROWS;

  for (int i = tid; i < HID * HID; i += 256) {           // Wgcn[k][c] -> sWT[c][k]
    int k = i >> 6, c = i & 63;
    sWT[c * LDK64 + k] = (_Float16)Wgcn[i];
  }
  if (tid < HID) spool[tid] = poolP[tid] / poolS[0];
  for (int i = tid; i < ROWS * HID; i += 256) {
    int r = i >> 6, c = i & 63;
    int gr = row0 + r;
    float v = 0.0f;
    if (gr < N) v = agg[(size_t)gr * HID + c] * rsqrtf(fmaxf(indeg[gr], 1.0f));
    sA[r * LDK64 + c] = (_Float16)v;
  }
  __syncthreads();

  const int tm = w >> 2, tn = w & 3;
  const int m0 = tm * 16, n0 = tn * 16;
  {
    v8f c = {};
#pragma unroll
    for (int k0 = 0; k0 < HID; k0 += 32)
      c = wmma_f16(frag_row(sA, LDK64, m0, k0, lane),
                   frag_row(sWT, LDK64, n0, k0, lane), c);
    int n  = n0 + (lane & 15);
    int mh = m0 + ((lane >> 4) << 3);
    float bias = bgcn[n];
#pragma unroll
    for (int r = 0; r < 8; ++r) sY[(mh + r) * HID + n] = c[r] + bias;
  }
  __syncthreads();

  // LayerNorm + residual concat: 8 lanes per row, 8 channels each
  const int row = tid >> 3, sub = tid & 7;
  const int gr  = row0 + row;
  float s1 = 0.f, s2 = 0.f;
#pragma unroll
  for (int j = 0; j < 8; ++j) {
    float v = sY[row * HID + sub * 8 + j];
    s1 += v; s2 += v * v;
  }
#pragma unroll
  for (int off = 1; off < 8; off <<= 1) {
    s1 += __shfl_xor(s1, off, 8);
    s2 += __shfl_xor(s2, off, 8);
  }
  float mean = s1 * (1.0f / HID);
  float var  = s2 * (1.0f / HID) - mean * mean;
  float inv  = rsqrtf(var + 1e-5f);
  if (gr < N) {
#pragma unroll
    for (int j = 0; j < 8; ++j) {
      int c = sub * 8 + j;
      float y  = sY[row * HID + c];
      float ln = (y - mean) * inv * ggcn[c] + begcn[c];
      float hv = h3[(size_t)gr * HID + c];
      x[(size_t)gr * W2 + c]       = ln - hv;        // loc_h
      x[(size_t)gr * W2 + HID + c] = spool[c] - hv;  // glo_h
    }
  }
}

// ---------------------------------------------------------------------------
// m = relu(x @ Wpool + bpool)  (per-node messages, 128x128 GEMM)
// ---------------------------------------------------------------------------
__global__ void __launch_bounds__(256) sage_msg_kernel(
    const float* __restrict__ x, const float* __restrict__ Wpool,
    const float* __restrict__ bpool, float* __restrict__ m, int N)
{
  __shared__ __align__(16) _Float16 sWT[W2 * LDK128];   // [n][k]
  __shared__ __align__(16) _Float16 sA [ROWS * LDK128]; // [m][k]
  const int tid = threadIdx.x, lane = tid & 31, w = tid >> 5;
  const int row0 = blockIdx.x * ROWS;

  for (int i = tid; i < W2 * W2; i += 256) {             // Wpool[k][c] -> sWT[c][k]
    int k = i >> 7, c = i & 127;
    sWT[c * LDK128 + k] = (_Float16)Wpool[i];
  }
  for (int i = tid; i < ROWS * W2; i += 256) {
    int r = i >> 7, c = i & 127;
    int gr = row0 + r;
    sA[r * LDK128 + c] = (gr < N) ? (_Float16)x[(size_t)gr * W2 + c] : (_Float16)0.0f;
  }
  __syncthreads();

  const int tm = w >> 2, m0 = tm * 16;
#pragma unroll
  for (int half = 0; half < 2; ++half) {
    int n0 = ((w & 3) + half * 4) * 16;
    v8f c = {};
#pragma unroll
    for (int k0 = 0; k0 < W2; k0 += 32)
      c = wmma_f16(frag_row(sA, LDK128, m0, k0, lane),
                   frag_row(sWT, LDK128, n0, k0, lane), c);
    int n  = n0 + (lane & 15);
    int mh = m0 + ((lane >> 4) << 3);
    float bias = bpool[n];
#pragma unroll
    for (int r = 0; r < 8; ++r) {
      int gr = row0 + mh + r;
      if (gr < N) m[(size_t)gr * W2 + n] = fmaxf(c[r] + bias, 0.0f);
    }
  }
}

// ---------------------------------------------------------------------------
// neigh[dst] = max(neigh[dst], m[src]); m >= 0 so int atomicMax on bits works
// ---------------------------------------------------------------------------
__global__ void neigh_max_kernel(const int* __restrict__ src, const int* __restrict__ dst,
                                 const float* __restrict__ m, float* __restrict__ neigh, int E) {
  int t = blockIdx.x * 256 + threadIdx.x;
  int e = t >> 5;            // 32 lanes per edge
  int c = (t & 31) * 4;
  if (e >= E) return;
  int s = src[e], d = dst[e];
  const float4 v = *(const float4*)&m[(size_t)s * W2 + c];
  int* a = (int*)&neigh[(size_t)d * W2 + c];
  atomicMax(a + 0, __float_as_int(v.x));
  atomicMax(a + 1, __float_as_int(v.y));
  atomicMax(a + 2, __float_as_int(v.z));
  atomicMax(a + 3, __float_as_int(v.w));
}

// ---------------------------------------------------------------------------
// x = relu(LN(x @ Wself + neigh @ Wneigh + b))   (in-place, row-local)
// One 34KB transposed-weight LDS buffer reused for Wself then Wneigh; f32
// accumulator fragments persist in VGPRs across the reload. sA/sN recycled
// as the f32 Y tile for the LayerNorm epilogue.
// ---------------------------------------------------------------------------
__global__ void __launch_bounds__(256) sage_update_kernel(
    const float* __restrict__ Wself, const float* __restrict__ Wneigh,
    const float* __restrict__ b, const float* __restrict__ g, const float* __restrict__ be,
    const float* __restrict__ neigh, float* __restrict__ x, int N)
{
  // layout: sWT (W2*LDK128 halves) | sA (ROWS*LDK128) | sN (ROWS*LDK128)
  __shared__ __align__(16) char smem[(W2 + 2 * ROWS) * LDK128 * 2];
  _Float16* sWT = (_Float16*)smem;
  _Float16* sA  = (_Float16*)(smem + W2 * LDK128 * 2);
  _Float16* sN  = (_Float16*)(smem + (W2 + ROWS) * LDK128 * 2);
  float*    sY  = (float*)   (smem + W2 * LDK128 * 2);   // aliases sA+sN after GEMMs

  const int tid = threadIdx.x, lane = tid & 31, w = tid >> 5;
  const int row0 = blockIdx.x * ROWS;

  for (int i = tid; i < ROWS * W2; i += 256) {
    int r = i >> 7, c = i & 127;
    int gr = row0 + r;
    sA[r * LDK128 + c] = (gr < N) ? (_Float16)x    [(size_t)gr * W2 + c] : (_Float16)0.0f;
    sN[r * LDK128 + c] = (gr < N) ? (_Float16)neigh[(size_t)gr * W2 + c] : (_Float16)0.0f;
  }
  for (int i = tid; i < W2 * W2; i += 256) {             // Wself[k][c] -> sWT[c][k]
    int k = i >> 7, c = i & 127;
    sWT[c * LDK128 + k] = (_Float16)Wself[i];
  }
  __syncthreads();

  const int tm  = w >> 2, m0 = tm * 16;
  const int n0a = (w & 3) * 16, n0b = n0a + 64;
  v8f c0 = {}, c1 = {};
#pragma unroll
  for (int k0 = 0; k0 < W2; k0 += 32) {
    v16h a = frag_row(sA, LDK128, m0, k0, lane);
    c0 = wmma_f16(a, frag_row(sWT, LDK128, n0a, k0, lane), c0);
    c1 = wmma_f16(a, frag_row(sWT, LDK128, n0b, k0, lane), c1);
  }
  __syncthreads();
  for (int i = tid; i < W2 * W2; i += 256) {             // Wneigh[k][c] -> sWT[c][k]
    int k = i >> 7, c = i & 127;
    sWT[c * LDK128 + k] = (_Float16)Wneigh[i];
  }
  __syncthreads();
#pragma unroll
  for (int k0 = 0; k0 < W2; k0 += 32) {
    v16h a = frag_row(sN, LDK128, m0, k0, lane);
    c0 = wmma_f16(a, frag_row(sWT, LDK128, n0a, k0, lane), c0);
    c1 = wmma_f16(a, frag_row(sWT, LDK128, n0b, k0, lane), c1);
  }
  __syncthreads();   // all reads of sA/sN done -> safe to recycle as sY

  {
    int n  = lane & 15;
    int mh = m0 + ((lane >> 4) << 3);
#pragma unroll
    for (int r = 0; r < 8; ++r) {
      sY[(mh + r) * W2 + n0a + n] = c0[r] + b[n0a + n];
      sY[(mh + r) * W2 + n0b + n] = c1[r] + b[n0b + n];
    }
  }
  __syncthreads();

  // LayerNorm + relu: 8 lanes per row, 16 channels each
  const int row = tid >> 3, sub = tid & 7;
  const int gr  = row0 + row;
  float s1 = 0.f, s2 = 0.f;
#pragma unroll
  for (int j = 0; j < 16; ++j) {
    float v = sY[row * W2 + sub * 16 + j];
    s1 += v; s2 += v * v;
  }
#pragma unroll
  for (int off = 1; off < 8; off <<= 1) {
    s1 += __shfl_xor(s1, off, 8);
    s2 += __shfl_xor(s2, off, 8);
  }
  float mean = s1 * (1.0f / W2);
  float var  = s2 * (1.0f / W2) - mean * mean;
  float inv  = rsqrtf(var + 1e-5f);
  if (gr < N) {
#pragma unroll
    for (int j = 0; j < 16; ++j) {
      int c = sub * 16 + j;
      float yv = sY[row * W2 + c];
      float ln = (yv - mean) * inv * g[c] + be[c];
      x[(size_t)gr * W2 + c] = fmaxf(ln, 0.0f);
    }
  }
}

// ---------------------------------------------------------------------------
// out = x @ W_out + b_out   (OUT=2: two lanes per row)
// ---------------------------------------------------------------------------
__global__ void out_kernel(const float* __restrict__ x, const float* __restrict__ Wout,
                           const float* __restrict__ bout, float* __restrict__ out, int N) {
  int t = blockIdx.x * 256 + threadIdx.x;
  int row = t >> 1, o = t & 1;
  if (row >= N) return;
  float acc = bout[o];
  const float* xr = &x[(size_t)row * W2];
  for (int c = 0; c < W2; ++c) acc += xr[c] * Wout[c * 2 + o];
  out[(size_t)row * 2 + o] = acc;
}

// ---------------------------------------------------------------------------
extern "C" void kernel_launch(void* const* d_in, const int* in_sizes, int n_in,
                              void* d_out, int out_size, void* d_ws, size_t ws_size,
                              hipStream_t stream) {
  (void)n_in; (void)out_size; (void)ws_size;
  const float* h     = (const float*)d_in[0];
  const int*   src   = (const int*)  d_in[1];
  const int*   dst   = (const int*)  d_in[2];
  const float* Win   = (const float*)d_in[3];
  const float* bin   = (const float*)d_in[4];
  const float* Wt1   = (const float*)d_in[5];
  const float* bt1   = (const float*)d_in[6];
  const float* Wt2   = (const float*)d_in[7];
  const float* bt2   = (const float*)d_in[8];
  const float* Wgcn  = (const float*)d_in[9];
  const float* bgcn  = (const float*)d_in[10];
  const float* ggcn  = (const float*)d_in[11];
  const float* begcn = (const float*)d_in[12];
  const float* Wgate = (const float*)d_in[13];
  const float* bgate = (const float*)d_in[14];
  const float* Wpool = (const float*)d_in[15];
  const float* bpool = (const float*)d_in[16];
  const float* Wself = (const float*)d_in[17];
  const float* Wneigh= (const float*)d_in[18];
  const float* sbv   = (const float*)d_in[19];
  const float* sgv   = (const float*)d_in[20];
  const float* sbev  = (const float*)d_in[21];
  const float* Wout  = (const float*)d_in[22];
  const float* bout  = (const float*)d_in[23];

  const int N = in_sizes[0] / IN_DIM;
  const int E = in_sizes[1];
  const int L = in_sizes[15] / (W2 * W2);

  // workspace layout (all float, packed, unpadded pieces)
  float* ws = (float*)d_ws;
  size_t off = 0;
  float* h3     = ws + off; off += (size_t)N * HID;
  float* logits = ws + off; off += (size_t)N;
  float* outdeg = ws + off; off += (size_t)N;   // outdeg/indeg/agg contiguous
  float* indeg  = ws + off; off += (size_t)N;
  float* agg    = ws + off; off += (size_t)N * HID;
  float* xbuf   = ws + off; off += (size_t)N * W2;
  float* mbuf   = ws + off; off += (size_t)N * W2;
  float* neigh  = ws + off; off += (size_t)N * W2;
  float* maxbuf = ws + off; off += 1;
  float* poolP  = ws + off; off += HID;         // poolP/poolS contiguous
  float* poolS  = ws + off; off += 1;

  const int nb32 = (N + ROWS - 1) / ROWS;

  // zero the accumulation buffers
  size_t zc = (size_t)N * (2 + HID);  // outdeg + indeg + agg
  fill_kernel<<<(int)((zc + 255) / 256), 256, 0, stream>>>(outdeg, zc, 0.0f);
  fill_kernel<<<1, 256, 0, stream>>>(poolP, (size_t)(HID + 1), 0.0f);

  fused_mlp_kernel<<<nb32, 256, 0, stream>>>(h, Win, bin, Wt1, bt1, Wt2, bt2,
                                             Wgate, bgate, h3, logits, N);
  degree_kernel<<<(E + 255) / 256, 256, 0, stream>>>(src, dst, outdeg, indeg, E);
  gcn_scatter_kernel<<<(E + 15) / 16, 256, 0, stream>>>(src, dst, h3, outdeg, agg, E);
  logit_max_kernel<<<1, 256, 0, stream>>>(logits, maxbuf, N);
  pool_partial_kernel<<<256, 256, 0, stream>>>(logits, h3, maxbuf, poolP, poolS, N);
  build_x_kernel<<<nb32, 256, 0, stream>>>(agg, indeg, Wgcn, bgcn, ggcn, begcn,
                                           h3, poolP, poolS, xbuf, N);

  for (int l = 0; l < L; ++l) {
    sage_msg_kernel<<<nb32, 256, 0, stream>>>(xbuf, Wpool + (size_t)l * W2 * W2,
                                              bpool + (size_t)l * W2, mbuf, N);
    fill_kernel<<<(int)(((size_t)N * W2 + 255) / 256), 256, 0, stream>>>(neigh, (size_t)N * W2, 0.0f);
    neigh_max_kernel<<<(E + 7) / 8, 256, 0, stream>>>(src, dst, mbuf, neigh, E);
    sage_update_kernel<<<nb32, 256, 0, stream>>>(Wself + (size_t)l * W2 * W2,
                                                 Wneigh + (size_t)l * W2 * W2,
                                                 sbv + (size_t)l * W2, sgv + (size_t)l * W2,
                                                 sbev + (size_t)l * W2, neigh, xbuf, N);
  }

  out_kernel<<<(2 * N + 255) / 256, 256, 0, stream>>>(xbuf, Wout, bout, (float*)d_out, N);
}
